// GraphEncoder_51960514347167
// MI455X (gfx1250) — matrix operand order repaired and checked by reading, not compiled, not run on hardware
//
#include <hip/hip_runtime.h>
#include <cstdint>

// ---------------------------------------------------------------------------
// Graph TransformerConv encoder for MI455X (gfx1250, wave32, WMMA).
// All dense GEMMs ([20000,256] x [256,256]) run through v_wmma_f32_16x16x32_bf16
// with f32 accumulation; A and B operands are pre-packed into the exact 16-bit
// WMMA VGPR fragment layouts so the GEMM inner loop is b128 loads + WMMA only.
// Each wave register-blocks a 32x64 C tile (2 M-tiles x 4 N-tiles): every B
// fragment is reused by two WMMAs, doubling math per load vs a 16x64 tile.
// Edge softmax/aggregation phases are L2-resident atomic passes.
// ---------------------------------------------------------------------------

#define NNODES  20000
#define NEDGES  320000
#define EMBED   256
#define HEADS   8
#define HEAD_DIM 32
#define NLAYERS 3
#define MTILES  (NNODES / 16)   // 1250
#define KTILES  (EMBED / 32)    // 8
#define NTILES  (EMBED / 16)    // 16
#define WMAT    65536           // elements per 256x256 weight matrix

typedef __bf16  bf16_t;
typedef bf16_t  v16bf __attribute__((ext_vector_type(16)));
typedef float   v8f   __attribute__((ext_vector_type(8)));

// ---------------------------------------------------------------------------
// Weight repack: W[k][n] f32 (row-major 256x256) -> B-fragment order bf16.
// B 32x16 16-bit layout: VGPR0..3 K=0..7, VGPR4..7 K=16..23 (lanes 0-15);
// lanes 16-31 add +8 to K, N = lane&15. One warp packs one (nt,kt) tile; each
// lane owns 16 contiguous bf16 (= its v16bf fragment).
// ---------------------------------------------------------------------------
__global__ __launch_bounds__(256) void pack_w_kernel(const float* __restrict__ W,
                                                     bf16_t* __restrict__ dst) {
    int warp = (blockIdx.x * 256 + threadIdx.x) >> 5;    // 0..127
    int lane = threadIdx.x & 31;
    int nt = warp / KTILES;
    int kt = warp % KTILES;
    int half = lane >> 4;
    int n    = lane & 15;
    bf16_t* out = dst + (((nt * KTILES + kt) * 32 + lane) << 4);
#pragma unroll
    for (int j = 0; j < 16; ++j) {
        int g = j >> 1, sub = j & 1;
        int k = (g < 4 ? 2 * g : 16 + 2 * (g - 4)) + sub + 8 * half;
        out[j] = (bf16_t)W[(kt * 32 + k) * EMBED + nt * 16 + n];
    }
}

// ---------------------------------------------------------------------------
// Activation repack: src[m][k] f32 -> A-fragment order bf16 (optional ReLU).
// A 16x32 16-bit layout (ISA 7.12.2): lanes 0-15 M=lane, VGPR0..3 K=0..7,
// VGPR4..7 K=16..23; lanes 16-31 add +8 to K. Each (mt,kt) tile = one warp.
// ---------------------------------------------------------------------------
template <bool RELU>
__global__ __launch_bounds__(256) void pack_a_kernel(const float* __restrict__ src,
                                                     bf16_t* __restrict__ dst) {
    int warp = (blockIdx.x * 256 + threadIdx.x) >> 5;    // 0..9999
    int lane = threadIdx.x & 31;
    int mt = warp / KTILES;
    int kt = warp % KTILES;
    int half = lane >> 4;
    int m    = lane & 15;
    const float* row = src + (size_t)(mt * 16 + m) * EMBED + kt * 32;
    bf16_t* out = dst + (((mt * KTILES + kt) * 32 + lane) << 4);
#pragma unroll
    for (int j = 0; j < 16; ++j) {
        int g = j >> 1, sub = j & 1;
        int k = (g < 4 ? 2 * g : 16 + 2 * (g - 4)) + sub + 8 * half;
        float f = row[k];
        if (RELU) f = fmaxf(f, 0.0f);
        out[j] = (bf16_t)f;
    }
}

// ---------------------------------------------------------------------------
// GEMM: C[M,256] = A[M,256] @ W[256,256] + bias, bf16 inputs, f32 accumulate.
// One wave computes a 32x64 C tile: per K-step, 2 A fragments + 4 B fragments
// feed 8 WMMAs (each B reused twice) -> 64 v_wmma_f32_16x16x32_bf16 per wave,
// 96 b128 loads. Grid: 625 blocks x 128 threads (4 waves) covering the
// 625 M-pairs x 4 N-groups exactly, so EXEC is all-ones for every WMMA.
// ---------------------------------------------------------------------------
__global__ __launch_bounds__(128) void gemm_bf16_wmma_kernel(const bf16_t* __restrict__ A,
                                                             const bf16_t* __restrict__ Bfrag,
                                                             const float* __restrict__ bias,
                                                             float* __restrict__ C) {
    int waveId = blockIdx.x * 4 + (threadIdx.x >> 5);    // 0..2499
    int lane   = threadIdx.x & 31;
    int mPair  = waveId >> 2;                            // 0..624
    int ntBase = (waveId & 3) * 4;                       // 0,4,8,12
    int mt0    = mPair * 2;
    int mt1    = mt0 + 1;

    const v16bf* Ap = (const v16bf*)A;
    const v16bf* Bp = (const v16bf*)Bfrag;

    v8f acc0[4], acc1[4];
#pragma unroll
    for (int i = 0; i < 4; ++i) {
        acc0[i] = (v8f){0.f, 0.f, 0.f, 0.f, 0.f, 0.f, 0.f, 0.f};
        acc1[i] = (v8f){0.f, 0.f, 0.f, 0.f, 0.f, 0.f, 0.f, 0.f};
    }

#pragma unroll
    for (int kt = 0; kt < KTILES; ++kt) {
        v16bf a0 = Ap[(mt0 * KTILES + kt) * 32 + lane];
        v16bf a1 = Ap[(mt1 * KTILES + kt) * 32 + lane];
#pragma unroll
        for (int i = 0; i < 4; ++i) {
            v16bf b = Bp[((ntBase + i) * KTILES + kt) * 32 + lane];
            acc0[i] = __builtin_amdgcn_wmma_f32_16x16x32_bf16(false, a0, false, b, (short)0, acc0[i], false, false);
            acc1[i] = __builtin_amdgcn_wmma_f32_16x16x32_bf16(false, a1, false, b, (short)0, acc1[i], false, false);
        }
    }

    // C/D layout: VGPR r -> row r (lanes 0-15) / row r+8 (lanes 16-31), col = lane&15.
    int half = lane >> 4;
    int ncol = lane & 15;
#pragma unroll
    for (int i = 0; i < 4; ++i) {
        int col = (ntBase + i) * 16 + ncol;
        float bv = bias[col];
#pragma unroll
        for (int r = 0; r < 8; ++r) {
            int row0 = mt0 * 16 + r + half * 8;
            int row1 = mt1 * 16 + r + half * 8;
            C[(size_t)row0 * EMBED + col] = acc0[i][r] + bv;
            C[(size_t)row1 * EMBED + col] = acc1[i][r] + bv;
        }
    }
}

// ---------------------------------------------------------------------------
// Edge-phase helpers: monotone float<->uint map for atomic segment-max.
// ---------------------------------------------------------------------------
__device__ __forceinline__ unsigned fmap_u(float f) {
    unsigned u = __float_as_uint(f);
    return (u & 0x80000000u) ? ~u : (u | 0x80000000u);
}
__device__ __forceinline__ float fumap(unsigned u) {
    return (u & 0x80000000u) ? __uint_as_float(u & 0x7FFFFFFFu) : __uint_as_float(~u);
}

__global__ __launch_bounds__(256) void seg_init_kernel(unsigned* __restrict__ mu,
                                                       float* __restrict__ den) {
    int t = blockIdx.x * 256 + threadIdx.x;   // exactly N*HEADS threads
    mu[t]  = 0u;                              // below mapped(-inf) in the ordering
    den[t] = 0.0f;
}

// logits[e,h] = <q[dst], k[src]>_head * scale ; fused atomic segment-max.
__global__ __launch_bounds__(256) void edge_logits_kernel(const int* __restrict__ src,
                                                          const int* __restrict__ dst,
                                                          const float* __restrict__ q,
                                                          const float* __restrict__ k,
                                                          float* __restrict__ logits,
                                                          unsigned* __restrict__ mu) {
    int t = blockIdx.x * 256 + threadIdx.x;   // exactly E*HEADS threads
    int e  = t >> 3;
    int hd = t & 7;
    int s = src[e], d = dst[e];
    const float4* qp = (const float4*)(q + (size_t)d * EMBED + hd * HEAD_DIM);
    const float4* kp = (const float4*)(k + (size_t)s * EMBED + hd * HEAD_DIM);
    float acc = 0.0f;
#pragma unroll
    for (int i = 0; i < 8; ++i) {
        float4 a = qp[i], b = kp[i];
        acc += a.x * b.x + a.y * b.y + a.z * b.z + a.w * b.w;
    }
    float lg = acc * 0.17677669529663687f;    // 1/sqrt(32)
    logits[t] = lg;
    atomicMax(mu + (size_t)d * HEADS + hd, fmap_u(lg));
}

// e = exp(logit - m[dst]); atomic segment-sum into denom; overwrite logits.
__global__ __launch_bounds__(256) void edge_exp_kernel(const int* __restrict__ dst,
                                                       const unsigned* __restrict__ mu,
                                                       float* __restrict__ logits,
                                                       float* __restrict__ den) {
    int t = blockIdx.x * 256 + threadIdx.x;
    int e  = t >> 3;
    int hd = t & 7;
    int d = dst[e];
    float m  = fumap(mu[(size_t)d * HEADS + hd]);
    float ex = __expf(logits[t] - m);
    logits[t] = ex;
    atomicAdd(den + (size_t)d * HEADS + hd, ex);
}

// alpha = e / denom[dst]  (denom > 0: this edge itself contributed).
__global__ __launch_bounds__(256) void edge_alpha_kernel(const int* __restrict__ dst,
                                                         const float* __restrict__ den,
                                                         float* __restrict__ logits) {
    int t = blockIdx.x * 256 + threadIdx.x;
    int e  = t >> 3;
    int hd = t & 7;
    logits[t] = logits[t] / den[(size_t)dst[e] * HEADS + hd];
}

// msg[dst] += alpha * v[src]  (msg pre-loaded with skip term h@Ws+bs by GEMM).
// Threads consecutive in channel -> coalesced v reads and atomic targets.
__global__ __launch_bounds__(256) void edge_scatter_kernel(const int* __restrict__ src,
                                                           const int* __restrict__ dst,
                                                           const float* __restrict__ alpha,
                                                           const float* __restrict__ v,
                                                           float* __restrict__ msg) {
    int t = blockIdx.x * 256 + threadIdx.x;   // exactly E*EMBED threads
    int e  = t >> 8;
    int c  = t & 255;
    int hd = c >> 5;
    int s = src[e], d = dst[e];
    float a = alpha[(size_t)e * HEADS + hd];
    atomicAdd(msg + (size_t)d * EMBED + c, a * v[(size_t)s * EMBED + c]);
}

// ---------------------------------------------------------------------------
// Host orchestration (graph-capture safe: only kernel launches on `stream`).
// ---------------------------------------------------------------------------
extern "C" void kernel_launch(void* const* d_in, const int* in_sizes, int n_in,
                              void* d_out, int out_size, void* d_ws, size_t ws_size,
                              hipStream_t stream) {
    (void)in_sizes; (void)n_in; (void)out_size; (void)ws_size;

    const float* x     = (const float*)d_in[0];
    const int*   ei    = (const int*)d_in[1];
    const float* W_in  = (const float*)d_in[2];
    const float* b_in  = (const float*)d_in[3];
    const float* W_out = (const float*)d_in[4];
    const float* b_out = (const float*)d_in[5];
    const float* Wq    = (const float*)d_in[6];
    const float* bq    = (const float*)d_in[7];
    const float* Wk    = (const float*)d_in[8];
    const float* bk    = (const float*)d_in[9];
    const float* Wv    = (const float*)d_in[10];
    const float* bv    = (const float*)d_in[11];
    const float* Wsk   = (const float*)d_in[12];
    const float* bsk   = (const float*)d_in[13];

    const int* srcI = ei;
    const int* dstI = ei + NEDGES;

    // Scratch carve-out (256B aligned slots).
    uintptr_t base = (uintptr_t)d_ws;
    size_t off = 0;
    auto take = [&](size_t bytes) -> void* {
        void* p = (void*)(base + off);
        off = (off + bytes + 255) & ~(size_t)255;
        return p;
    };
    bf16_t*   Afrag = (bf16_t*)take((size_t)NNODES * EMBED * sizeof(bf16_t));
    bf16_t*   Wfrag = (bf16_t*)take((size_t)14 * WMAT * sizeof(bf16_t));
    float*    qb    = (float*)take((size_t)NNODES * EMBED * sizeof(float));
    float*    kb    = (float*)take((size_t)NNODES * EMBED * sizeof(float));
    float*    vb    = (float*)take((size_t)NNODES * EMBED * sizeof(float));
    float*    msg   = (float*)take((size_t)NNODES * EMBED * sizeof(float));
    float*    lg    = (float*)take((size_t)NEDGES * HEADS * sizeof(float));
    unsigned* mu    = (unsigned*)take((size_t)NNODES * HEADS * sizeof(unsigned));
    float*    den   = (float*)take((size_t)NNODES * HEADS * sizeof(float));

    // 1) Repack all 14 weight matrices into WMMA B-fragment order (bf16).
    const float* wsrc[14] = {
        W_in, W_out,
        Wq, Wq + WMAT, Wq + 2 * WMAT,
        Wk, Wk + WMAT, Wk + 2 * WMAT,
        Wv, Wv + WMAT, Wv + 2 * WMAT,
        Wsk, Wsk + WMAT, Wsk + 2 * WMAT};
    for (int i = 0; i < 14; ++i)
        pack_w_kernel<<<dim3(NTILES * KTILES / 8), 256, 0, stream>>>(wsrc[i], Wfrag + (size_t)i * WMAT);

    const dim3 packGrid(MTILES * KTILES / 8);   // 1250 blocks
    const dim3 gemmGrid((MTILES / 2) * 4 / 4);  // 625 blocks, 4 waves each

    // 2) Input projection: h0 = x @ W_in + b_in
    pack_a_kernel<false><<<packGrid, 256, 0, stream>>>(x, Afrag);
    gemm_bf16_wmma_kernel<<<gemmGrid, 128, 0, stream>>>(Afrag, Wfrag + 0 * WMAT, b_in, msg);
    pack_a_kernel<false><<<packGrid, 256, 0, stream>>>(msg, Afrag);   // h0 -> A fragments

    // 3) Three TransformerConv layers.
    for (int l = 0; l < NLAYERS; ++l) {
        gemm_bf16_wmma_kernel<<<gemmGrid, 128, 0, stream>>>(Afrag, Wfrag + (size_t)(2 + l) * WMAT, bq + l * EMBED, qb);
        gemm_bf16_wmma_kernel<<<gemmGrid, 128, 0, stream>>>(Afrag, Wfrag + (size_t)(5 + l) * WMAT, bk + l * EMBED, kb);
        gemm_bf16_wmma_kernel<<<gemmGrid, 128, 0, stream>>>(Afrag, Wfrag + (size_t)(8 + l) * WMAT, bv + l * EMBED, vb);

        seg_init_kernel<<<dim3(NNODES * HEADS / 256), 256, 0, stream>>>(mu, den);
        edge_logits_kernel<<<dim3(NEDGES * HEADS / 256), 256, 0, stream>>>(srcI, dstI, qb, kb, lg, mu);
        edge_exp_kernel<<<dim3(NEDGES * HEADS / 256), 256, 0, stream>>>(dstI, mu, lg, den);
        edge_alpha_kernel<<<dim3(NEDGES * HEADS / 256), 256, 0, stream>>>(dstI, den, lg);

        // Skip term into msg, then scatter attention-weighted values on top.
        gemm_bf16_wmma_kernel<<<gemmGrid, 128, 0, stream>>>(Afrag, Wfrag + (size_t)(11 + l) * WMAT, bsk + l * EMBED, msg);
        edge_scatter_kernel<<<dim3(NEDGES), 256, 0, stream>>>(srcI, dstI, lg, vb, msg);

        // h = relu(msg) -> next layer's A fragments.
        pack_a_kernel<true><<<packGrid, 256, 0, stream>>>(msg, Afrag);
    }

    // 4) Output projection: out = h @ W_out + b_out
    gemm_bf16_wmma_kernel<<<gemmGrid, 128, 0, stream>>>(Afrag, Wfrag + 1 * WMAT, b_out, (float*)d_out);
}
